// HBMA_Optimized_27565100106067
// MI455X (gfx1250) — matrix operand order repaired and checked by reading, not compiled, not run on hardware
//
#include <hip/hip_runtime.h>
#include <stdint.h>

#define GH   135
#define GW   240
#define GHP  143
#define GWP  256
#define KDIM 768
#define NCH  12      // N*C
#define IMH  1080
#define IMW  1920
#define DSP  4
#define SS   9

typedef __attribute__((ext_vector_type(16))) __bf16 v16bf;
typedef __attribute__((ext_vector_type(8)))  float  v8f;

union FragU { uint4 q[2]; v16bf v; };

// Kernel 1: zero padded target-block array, +inf padded norms.
__global__ __launch_bounds__(256) void init_pad(uint4* __restrict__ tpad4,
                                                float* __restrict__ tnorm) {
  size_t i = (size_t)blockIdx.x * 256 + threadIdx.x;
  const size_t n4 = (size_t)GHP * GWP * KDIM * 2 / 16;
  if (i < n4) {
    uint4 z; z.x = z.y = z.z = z.w = 0u;
    tpad4[i] = z;
  }
  if (i < (size_t)GHP * GWP) tnorm[i] = __builtin_inff();
}

// Kernel 2: blockify to packed bf16 vectors + per-block |t|^2.
__global__ __launch_bounds__(256) void prep(const float* __restrict__ anchor,
                                            const float* __restrict__ target,
                                            __bf16* __restrict__ abf,
                                            __bf16* __restrict__ tpad,
                                            float* __restrict__ tnorm) {
  int b = blockIdx.x;
  int gy = b / GW, gx = b - gy * GW;
  int tid = threadIdx.x;
  size_t abase = ((size_t)gy * GW + gx) * KDIM;
  size_t tbase = ((size_t)(gy + DSP) * GWP + (gx + DSP)) * KDIM;
  float acc = 0.f;
  for (int e = tid; e < KDIM; e += 256) {
    int j = e & 7, i = (e >> 3) & 7, nc = e >> 6;
    size_t off = ((size_t)nc * IMH + gy * 8 + i) * IMW + gx * 8 + j;
    float a = anchor[off], t = target[off];
    abf[abase + e] = (__bf16)a;
    __bf16 tb = (__bf16)t;
    tpad[tbase + e] = tb;
    float tf = (float)tb;
    acc += tf * tf;
  }
  __shared__ float red[256];
  red[tid] = acc;
  __syncthreads();
  for (int s = 128; s > 0; s >>= 1) {
    if (tid < s) red[tid] += red[tid + s];
    __syncthreads();
  }
  if (tid == 0) tnorm[(size_t)(gy + DSP) * GWP + gx + DSP] = red[0];
}

// Kernel 3: banded cross-correlation via v_wmma_f32_16x16x32_bf16 + argmin.
__global__ __launch_bounds__(288) void hbma_wmma(const __bf16* __restrict__ abf,
                                                 const __bf16* __restrict__ tpad,
                                                 const float*  __restrict__ tnorm,
                                                 int* __restrict__ bidx) {
  __shared__ __attribute__((aligned(16))) __bf16 Alds[16 * KDIM];
  __shared__ float Clds[SS * 2 * 256];
  __shared__ float redC[160];
  __shared__ int   redK[160];

  int gx0 = blockIdx.x * 16;
  int gy  = blockIdx.y;
  int tid = threadIdx.x;

  { // stage A tile (contiguous 24576 B) into LDS
    const uint4* src = (const uint4*)(abf + ((size_t)gy * GW + gx0) * KDIM);
    uint4* dst = (uint4*)Alds;
    for (int i = tid; i < 16 * KDIM * 2 / 16; i += 288) dst[i] = src[i];
  }
  __syncthreads();

  int wave = tid >> 5;          // dy slot 0..8
  int lane = tid & 31;
  int r = lane & 15, hi = lane >> 4;
  int gyp = gy + wave;          // padded target row

  const __bf16* b0p = tpad + ((size_t)gyp * GWP + gx0 + r) * KDIM + hi * 16;
  const __bf16* b1p = b0p + (size_t)16 * KDIM;
  const __bf16* a0  = Alds + r * KDIM + hi * 16;

  v8f c0 = {}, c1 = {};
  #pragma unroll 4
  for (int k0 = 0; k0 < KDIM; k0 += 32) {
    FragU a, b0, b1;
    a.q[0]  = *(const uint4*)(a0  + k0);
    a.q[1]  = *(const uint4*)(a0  + k0 + 8);
    b0.q[0] = *(const uint4*)(b0p + k0);
    b0.q[1] = *(const uint4*)(b0p + k0 + 8);
    b1.q[0] = *(const uint4*)(b1p + k0);
    b1.q[1] = *(const uint4*)(b1p + k0 + 8);
    c0 = __builtin_amdgcn_wmma_f32_16x16x32_bf16(false, a.v, false, b0.v,
                                                 (short)0, c0, false, false);
    c1 = __builtin_amdgcn_wmma_f32_16x16x32_bf16(false, a.v, false, b1.v,
                                                 (short)0, c1, false, false);
  }

  {
    float* base = Clds + (wave * 2) * 256;
    #pragma unroll
    for (int i = 0; i < 8; i++) {
      base[i * 32 + lane]       = c0[i];
      base[256 + i * 32 + lane] = c1[i];
    }
  }
  __syncthreads();

  if (tid < 144) {
    int m = tid & 15, dyI = tid >> 4;
    int gyp2 = gy + dyI;
    float best = __builtin_inff();
    int bk = 0;
    #pragma unroll
    for (int dxI = 0; dxI < SS; dxI++) {
      int jg = m + dxI;
      int tile = jg >> 4, j = jg & 15;
      float cross = Clds[((dyI * 2 + tile) * 8 + (m & 7)) * 32 +
                         ((m >> 3) << 4) + j];
      float tn = tnorm[(size_t)gyp2 * GWP + gx0 + jg];
      float cost = tn - 2.0f * cross;
      int k = dyI * SS + dxI;
      if (cost < best) { best = cost; bk = k; }
    }
    redC[tid] = best; redK[tid] = bk;
  }
  __syncthreads();
  if (tid < 16) {
    float best = __builtin_inff();
    int bk = 0;
    for (int d = 0; d < SS; d++) {
      float c = redC[d * 16 + tid];
      if (c < best) { best = c; bk = redK[d * 16 + tid]; }
    }
    bidx[gy * GW + gx0 + tid] = bk;
  }
}

// Kernel 4: coalesced float4 gather of winning anchor blocks.
__global__ __launch_bounds__(256) void gather_out(const float* __restrict__ anchor,
                                                  const int* __restrict__ bidx,
                                                  float* __restrict__ out) {
  int v = blockIdx.x * 256 + threadIdx.x;
  const int total = NCH * IMH * (IMW / 4);
  if (v >= total) return;
  int xq = v % (IMW / 4);
  int rest = v / (IMW / 4);
  int y = rest % IMH;
  int nc = rest / IMH;
  int x = xq * 4;
  int gy = y >> 3, gx = x >> 3;
  int k = bidx[gy * GW + gx];
  int sy = y + (k / SS - DSP) * 8;
  int sx = x + (k % SS - DSP) * 8;
  const float4* src = (const float4*)(anchor + ((size_t)nc * IMH + sy) * IMW + sx);
  float4* dst = (float4*)(out + ((size_t)nc * IMH + y) * IMW + x);
  *dst = *src;
}

extern "C" void kernel_launch(void* const* d_in, const int* in_sizes, int n_in,
                              void* d_out, int out_size, void* d_ws, size_t ws_size,
                              hipStream_t stream) {
  (void)in_sizes; (void)n_in; (void)out_size; (void)ws_size;
  const float* anchor = (const float*)d_in[0];
  const float* target = (const float*)d_in[1];
  float* out = (float*)d_out;

  char* ws = (char*)d_ws;
  size_t szA = (size_t)GH * GW * KDIM * 2;
  size_t szT = (size_t)GHP * GWP * KDIM * 2;
  size_t szN = (size_t)GHP * GWP * 4;
  size_t offA = 0;
  size_t offT = offA + szA;
  size_t offN = offT + szT;
  size_t offI = offN + ((szN + 255) & ~(size_t)255);

  __bf16* abf   = (__bf16*)(ws + offA);
  __bf16* tpad  = (__bf16*)(ws + offT);
  float*  tnorm = (float*)(ws + offN);
  int*    bidx  = (int*)(ws + offI);

  {
    size_t n4 = szT / 16;
    int blocks = (int)((n4 + 255) / 256);
    init_pad<<<blocks, 256, 0, stream>>>((uint4*)tpad, tnorm);
  }
  prep<<<GH * GW, 256, 0, stream>>>(anchor, target, abf, tpad, tnorm);

  dim3 grid(GW / 16, GH);
  hbma_wmma<<<grid, 288, 0, stream>>>(abf, tpad, tnorm, bidx);

  int total4 = NCH * IMH * (IMW / 4);
  gather_out<<<(total4 + 255) / 256, 256, 0, stream>>>(anchor, bidx, out);
}
